// LogReg_73315091742880
// MI455X (gfx1250) — matrix operand order repaired and checked by reading, not compiled, not run on hardware
//
#include <hip/hip_runtime.h>

typedef __attribute__((ext_vector_type(16))) _Float16 v16h;
typedef __attribute__((ext_vector_type(8)))  float    v8f;

#define K1   256
#define N1   128
#define K2   128
#define NCLS 10

union Frag { unsigned u[8]; v16h v; };

static __device__ inline unsigned pack2f16(float a, float b) {
    union { _Float16 h[2]; unsigned u; } p;
    p.h[0] = (_Float16)a;
    p.h[1] = (_Float16)b;
    return p.u;
}

__global__ __launch_bounds__(256) void fused_ln_mlp_kernel(
    const float* __restrict__ seq,
    const float* __restrict__ ln_g,
    const float* __restrict__ ln_b,
    const float* __restrict__ fc_w,
    const float* __restrict__ fc_b,
    const float* __restrict__ mlp_w,
    const float* __restrict__ mlp_b,
    float* __restrict__ out)
{
    __shared__ float     xs[16 * K1];     // raw f32 tile        (16 KB)
    __shared__ _Float16  aT[16 * K1];     // normalized f16 tile ( 8 KB)
    __shared__ _Float16  hT[16 * K2];     // sigmoid(fc) f16     ( 4 KB)
    __shared__ float     red_s[256];
    __shared__ float     red_q[256];
    __shared__ float     mu[16];
    __shared__ float     rstd[16];

    const int tid     = threadIdx.x;
    const int rowBase = blockIdx.x * 16;          // 88 blocks * 16 rows = 1408

    // hint the (tiny, reused-by-every-block) weight matrices into cache
    __builtin_prefetch(fc_w  + (tid & 127) * K1, 0, 1);
    __builtin_prefetch(mlp_w + (tid & 7)   * K2, 0, 1);

    // ---- 1) async-load 16x256 f32 tile straight into LDS (no VGPR round-trip)
    //      4 x b128 per thread; per-lane LDS byte address = low 32 bits of the
    //      flat shared pointer (aperture mapping truncates addr[31:0]).
    {
        const unsigned ldsBase = (unsigned)(size_t)(void*)xs + (unsigned)tid * 16u;
        const float*   gp      = seq + (size_t)rowBase * K1 + tid * 4;
        #pragma unroll
        for (int i = 0; i < 4; ++i) {
            unsigned     la = ldsBase + (unsigned)i * 4096u;   // 256 thr * 16B
            const void*  ga = (const void*)(gp + i * 1024);    // 256 thr * 4 floats
            asm volatile("global_load_async_to_lds_b128 %0, %1, off"
                         :: "v"(la), "v"(ga) : "memory");
        }
        asm volatile("s_wait_asynccnt 0" ::: "memory");
    }
    __syncthreads();

    // ---- 2) LayerNorm statistics: 16 threads per row ----
    const int r = tid >> 4;       // row 0..15
    const int c = tid & 15;       // sub-column 0..15
    {
        float s = 0.f, q = 0.f;
        #pragma unroll
        for (int j = 0; j < 16; ++j) {
            float v = xs[r * K1 + c + 16 * j];
            s += v; q += v * v;
        }
        red_s[tid] = s; red_q[tid] = q;
    }
    __syncthreads();
    if (tid < 16) {
        float ss = 0.f, qq = 0.f;
        #pragma unroll
        for (int j = 0; j < 16; ++j) { ss += red_s[tid * 16 + j]; qq += red_q[tid * 16 + j]; }
        float m   = ss * (1.0f / 256.0f);
        float var = qq * (1.0f / 256.0f) - m * m;
        mu[tid]   = m;
        rstd[tid] = rsqrtf(var + 1e-5f);
    }
    __syncthreads();

    // ---- 3) normalize + affine -> f16 A tile (row-major in LDS) ----
    {
        float m = mu[r], k = rstd[r];
        #pragma unroll
        for (int j = 0; j < 16; ++j) {
            int col = c + 16 * j;
            float v = (xs[r * K1 + col] - m) * k * ln_g[col] + ln_b[col];
            aT[r * K1 + col] = (_Float16)v;
        }
    }
    __syncthreads();

    const int wave = tid >> 5;    // 0..7 -> fc column tile
    const int lane = tid & 31;
    const int ln16 = lane & 15;
    const int lhi  = lane >> 4;

    // ---- 4) GEMM1: h = sigmoid(A[16x256] x fc_w^T[256x128] + b), one N-tile per wave
    v8f acc = {};
    const int nBase = wave * 16;
    const float* wrow = fc_w + (size_t)(nBase + ln16) * K1;   // B col n -> fc_w row n
    #pragma unroll
    for (int s = 0; s < 8; ++s) {
        const int kb = 32 * s;
        Frag fa, fb;
        // A fragment (16-bit A 16x32 layout): lane row = ln16
        #pragma unroll
        for (int j = 0; j < 8; ++j) {
            int k = kb + ((j < 4) ? (8 * lhi + 2 * j) : (16 + 8 * lhi + 2 * (j - 4)));
            fa.u[j] = *(const unsigned*)(&aT[ln16 * K1 + k]);
        }
        // B fragment (32x16): lane col = ln16; V_j holds K = 8*lhi + 2j (j<4) / 16 + 8*lhi + 2(j-4), +1
        #pragma unroll
        for (int j = 0; j < 8; ++j) {
            int k = kb + ((j < 4) ? (8 * lhi + 2 * j) : (16 + 8 * lhi + 2 * (j - 4)));
            fb.u[j] = pack2f16(wrow[k], wrow[k + 1]);
        }
        acc = __builtin_amdgcn_wmma_f32_16x16x32_f16(
                  false, fa.v, false, fb.v, (short)0, acc, false, false);
    }

    // ---- 5) bias + sigmoid, scatter to LDS per C/D layout: acc[j] -> row 8*lhi+j, col nBase+ln16
    {
        float b = fc_b[nBase + ln16];
        #pragma unroll
        for (int j = 0; j < 8; ++j) {
            float v = acc[j] + b;
            v = 1.0f / (1.0f + __expf(-v));
            hT[(8 * lhi + j) * K2 + nBase + ln16] = (_Float16)v;
        }
    }
    __syncthreads();

    // ---- 6) GEMM2: out = h[16x128] x mlp_w^T[128x16pad] + b  (wave 0, EXEC all-1s in-wave)
    if (wave == 0) {
        v8f acc2 = {};
        const float* w2 = mlp_w + (size_t)ln16 * K2;   // valid only for ln16 < 10
        #pragma unroll
        for (int s = 0; s < 4; ++s) {
            const int kb = 32 * s;
            Frag fa, fb;
            #pragma unroll
            for (int j = 0; j < 8; ++j) {
                int k = kb + ((j < 4) ? (8 * lhi + 2 * j) : (16 + 8 * lhi + 2 * (j - 4)));
                fa.u[j] = *(const unsigned*)(&hT[ln16 * K2 + k]);
            }
            #pragma unroll
            for (int j = 0; j < 8; ++j) {
                unsigned u = 0u;                       // zero-pad columns 10..15
                if (ln16 < NCLS) {
                    int k = kb + ((j < 4) ? (8 * lhi + 2 * j) : (16 + 8 * lhi + 2 * (j - 4)));
                    u = pack2f16(w2[k], w2[k + 1]);
                }
                fb.u[j] = u;
            }
            acc2 = __builtin_amdgcn_wmma_f32_16x16x32_f16(
                       false, fa.v, false, fb.v, (short)0, acc2, false, false);
        }
        if (ln16 < NCLS) {
            float b = mlp_b[ln16];
            #pragma unroll
            for (int j = 0; j < 8; ++j) {
                int row = rowBase + 8 * lhi + j;       // 1408 = 88*16, always in range
                out[(size_t)row * NCLS + ln16] = acc2[j] + b;
            }
        }
    }
}

extern "C" void kernel_launch(void* const* d_in, const int* in_sizes, int n_in,
                              void* d_out, int out_size, void* d_ws, size_t ws_size,
                              hipStream_t stream) {
    (void)in_sizes; (void)n_in; (void)d_ws; (void)ws_size; (void)out_size;
    const float* seq   = (const float*)d_in[0];
    const float* ln_g  = (const float*)d_in[1];
    const float* ln_b  = (const float*)d_in[2];
    const float* fc_w  = (const float*)d_in[3];
    const float* fc_b  = (const float*)d_in[4];
    const float* mlp_w = (const float*)d_in[5];
    const float* mlp_b = (const float*)d_in[6];
    float* out = (float*)d_out;

    fused_ln_mlp_kernel<<<88, 256, 0, stream>>>(seq, ln_g, ln_b, fc_w, fc_b,
                                                mlp_w, mlp_b, out);
}
